// MultiHeadAttention_43671227466060
// MI455X (gfx1250) — compile-verified, matrix operand order checked by reading, not run on hardware
//
#include <hip/hip_runtime.h>
#include <hip/hip_bf16.h>

// ---------------------------------------------------------------------------
// MultiHeadAttention forward for MI455X (gfx1250), wave32 + WMMA f16/f32,
// Tensor Data Mover (TDM) staging for flash-attention K/V tiles.
// B=64, T=256, C=384, H=6, D=64.
// ---------------------------------------------------------------------------

typedef __attribute__((ext_vector_type(16))) _Float16 v16h;
typedef __attribute__((ext_vector_type(8)))  _Float16 h8v;
typedef __attribute__((ext_vector_type(8)))  float    v8f;
typedef __attribute__((ext_vector_type(4)))  unsigned int u32x4;
typedef __attribute__((ext_vector_type(4)))  int      i32x4;
typedef __attribute__((ext_vector_type(8)))  int      i32x8;

static constexpr int kB = 64;
static constexpr int kT = 256;
static constexpr int kC = 384;
static constexpr int kH = 6;
static constexpr int kD = 64;
static constexpr int kBT = kB * kT;      // 16384 rows
static constexpr int kKB = 128;          // keys per staged block

#if defined(__HIP_DEVICE_COMPILE__) && __has_builtin(__builtin_amdgcn_tensor_load_to_lds) && __has_builtin(__builtin_amdgcn_s_wait_tensorcnt)
#define USE_TDM 1
#else
#define USE_TDM 0
#endif

// 16x16x32 f16 WMMA, f32 accumulate
static __device__ __forceinline__ v8f wmma_f16(v16h a, v16h b, v8f c) {
  return __builtin_amdgcn_wmma_f32_16x16x32_f16(
      false, a, false, b, (short)0, c, false, false);
}

// Load a 16x32 f16 fragment (A layout; also used for B via n<->m symmetry).
// Element h of lane = src[(lane&15)*ld + (h&7) + 8*(lane>>4) + 16*(h>>3)].
// Requires src 16B-aligned and ld a multiple of 8 halfs.
template <typename PtrT>
static __device__ __forceinline__ v16h load_frag(PtrT src, int ld, int lane) {
  const int row = lane & 15;
  const int hi8 = (lane >> 4) * 8;
  union { v16h v; h8v p[2]; } u;
  u.p[0] = *(const h8v*)(src + row * ld + hi8);
  u.p[1] = *(const h8v*)(src + row * ld + hi8 + 16);
  return u.v;
}

// B fragment from row-major K x N storage (k-major rows): element h of lane =
// src[((h&7) + 8*(lane>>4) + 16*(h>>3)) * ld + (lane&15)].
// Strided 2-byte loads; lanes 0..15 touch consecutive bytes (conflict-free).
template <typename PtrT>
static __device__ __forceinline__ v16h load_frag_kmajor(PtrT src, int ld, int lane) {
  const PtrT p = src + (lane & 15) + (lane >> 4) * 8 * ld;
  v16h v;
#pragma unroll
  for (int h = 0; h < 16; ++h) v[h] = p[((h & 7) + 16 * (h >> 3)) * ld];
  return v;
}

#if USE_TDM
// Issue a TDM 2D tile load (f16 elements) global -> LDS.
// dim0 = contiguous elements per row (tile & tensor), rows = tile rows,
// stride0 = row stride in elements. D# layout per CDNA5 ISA.
static __device__ __forceinline__ void tdm_load_to_lds_2d(
    unsigned int lds_byte_addr, const void* gsrc,
    unsigned int dim0, unsigned int rows, unsigned int stride0) {
  const unsigned long long ga = (unsigned long long)(size_t)gsrc;
  u32x4 g0;
  g0[0] = 1u;                                   // count=1, user descriptor
  g0[1] = lds_byte_addr;                        // D#.lds_addr
  g0[2] = (unsigned int)ga;                     // D#.global_addr[31:0]
  g0[3] = (unsigned int)((ga >> 32) & 0x01FFFFFFu) | (2u << 30);  // addr hi | type=2
  i32x8 g1;
  g1[0] = 0x10000;                              // data_size=1 (2 bytes), no flags
  g1[1] = (int)((dim0 & 0xFFFFu) << 16);        // tensor_dim0[15:0]
  g1[2] = (int)((dim0 >> 16) | ((rows & 0xFFFFu) << 16));  // dim0 hi | tensor_dim1 lo
  g1[3] = (int)((rows >> 16) | ((dim0 & 0xFFFFu) << 16));  // dim1 hi | tile_dim0
  g1[4] = (int)(rows & 0xFFFFu);                // tile_dim1
  g1[5] = (int)stride0;                         // tensor_dim0_stride[31:0]
  g1[6] = 0;
  g1[7] = 0;
  i32x4 g2 = {0, 0, 0, 0};
  i32x4 g3 = {0, 0, 0, 0};
#if defined(__clang_major__) && __clang_major__ >= 23
  i32x8 g4 = {0, 0, 0, 0, 0, 0, 0, 0};
  __builtin_amdgcn_tensor_load_to_lds(g0, g1, g2, g3, g4, 0);
#else
  __builtin_amdgcn_tensor_load_to_lds(g0, g1, g2, g3, 0);
#endif
}
#endif  // USE_TDM

// ---------------------------------------------------------------------------
// Kernel 1: fp32 -> f16 weight conversion, packed [Wk | Wq | Wv | Wo]
// ---------------------------------------------------------------------------
__global__ __launch_bounds__(256) void cvt_weights_kernel(
    const float* __restrict__ Wk, const float* __restrict__ Wq,
    const float* __restrict__ Wv, const float* __restrict__ Wo,
    _Float16* __restrict__ Wh) {
  int i = blockIdx.x * 256 + threadIdx.x;
  if (i >= 4 * kC * kC) return;
  int mat = i / (kC * kC);
  int r = i - mat * (kC * kC);
  const float* src = (mat == 0) ? Wk : (mat == 1) ? Wq : (mat == 2) ? Wv : Wo;
  Wh[i] = (_Float16)src[r];
}

// ---------------------------------------------------------------------------
// Kernel 2: fused QKV projection. Block = 16 rows of x. 8 waves; each wave
// owns 3 column tiles per matrix (Q,K,V). B fragments are batch-loaded per
// K-chunk so the 18 global_load_b128 clause together ahead of the 9 WMMAs.
// ---------------------------------------------------------------------------
__global__ __launch_bounds__(256) void qkv_kernel(
    const float* __restrict__ x, const _Float16* __restrict__ Wh,
    _Float16* __restrict__ Qh, _Float16* __restrict__ Kh,
    _Float16* __restrict__ Vh) {
  __shared__ __align__(16) _Float16 Xs[16][kC];  // 12 KB

  const int tid = threadIdx.x;
  const int row0 = blockIdx.x * 16;

  // Stage + convert x tile (coalesced fp32 reads, f16 LDS stores)
  const float* xsrc = x + (size_t)row0 * kC;
#pragma unroll
  for (int i = 0; i < 24; ++i) {
    int e = tid + i * 256;  // 0..6143
    Xs[e / kC][e % kC] = (_Float16)xsrc[e];
  }
  __syncthreads();

  const int w = tid >> 5, lane = tid & 31;
  const int col = lane & 15, hi = lane >> 4;

  v8f acc[9] = {};  // [mat*3 + s], mat 0=Q 1=K 2=V
#pragma unroll
  for (int kc = 0; kc < 12; ++kc) {
    v16h a = load_frag(&Xs[0][kc * 32], kC, lane);
    v16h bf[9];
#pragma unroll
    for (int t9 = 0; t9 < 9; ++t9) {
      const int mat = t9 / 3;
      const int wsel = (mat == 0) ? 1 : (mat == 1) ? 0 : 2;  // Wh order k,q,v,o
      const int j0 = (w * 3 + (t9 % 3)) * 16;
      bf[t9] = load_frag(Wh + wsel * kC * kC + j0 * kC + kc * 32, kC, lane);
    }
#pragma unroll
    for (int t9 = 0; t9 < 9; ++t9) acc[t9] = wmma_f16(a, bf[t9], acc[t9]);
  }

  const int b = row0 / kT, t0 = row0 % kT;
#pragma unroll
  for (int t9 = 0; t9 < 9; ++t9) {
    const int mat = t9 / 3;
    _Float16* dst = (mat == 0) ? Qh : (mat == 1) ? Kh : Vh;
    const int j0 = (w * 3 + (t9 % 3)) * 16;
    const int hd = j0 / kD, d0 = j0 % kD;
#pragma unroll
    for (int r = 0; r < 8; ++r) {
      const int t = t0 + r + 8 * hi;
      dst[((size_t)(b * kH + hd) * kT + t) * kD + d0 + col] = (_Float16)acc[t9][r];
    }
  }
}

// ---------------------------------------------------------------------------
// Kernel 3: flash attention. Block = (b, head, 128-query half). 8 waves, each
// owns a 16-row q-tile. K/V blocks DMA'd into LDS via TDM (tensor_load_to_lds
// + s_wait_tensorcnt); online softmax with lane-group shuffle reductions.
// ---------------------------------------------------------------------------
__global__ __launch_bounds__(256) void attn_kernel(
    const _Float16* __restrict__ Qh, const _Float16* __restrict__ Kh,
    const _Float16* __restrict__ Vh, _Float16* __restrict__ Oh) {
  __shared__ __align__(16) _Float16 Ks[kKB][kD];      // 16 KB
  __shared__ __align__(16) _Float16 Vs[kKB][kD];      // 16 KB (row-major)
  __shared__ __align__(16) _Float16 Ps[8][16][40];    // 10 KB per-wave P scratch

  const int tid = threadIdx.x, w = tid >> 5, lane = tid & 31;
  const int col = lane & 15, hi = lane >> 4;
  const int bh = blockIdx.x >> 1, qhalf = blockIdx.x & 1;
  const int b = bh / kH, hd = bh % kH;

  const size_t head_off = (size_t)(b * kH + hd) * kT * kD;
  const int q0 = qhalf * 128 + w * 16;  // first query row of this wave

  // Q fragments for d=0..31 and d=32..63 (stay in VGPRs for whole kernel)
  const _Float16* qbase = Qh + head_off + (size_t)q0 * kD;
  const v16h aq0 = load_frag(qbase, kD, lane);
  const v16h aq1 = load_frag(qbase + 32, kD, lane);

  float m[8], l[8];
  v8f o[4] = {};
#pragma unroll
  for (int r = 0; r < 8; ++r) { m[r] = -1e30f; l[r] = 0.f; }
  const float sc = 0.05103103630798288f;  // 384^-0.5

  for (int kb = 0; kb <= qhalf; ++kb) {
    if (kb) __syncthreads();
    const _Float16* ksrc = Kh + head_off + (size_t)(kb * kKB) * kD;
    const _Float16* vsrc = Vh + head_off + (size_t)(kb * kKB) * kD;
#if USE_TDM
    // Tensor Data Mover: DMA the 128x64 f16 K and V tiles into LDS.
    if (w == 0) {
      tdm_load_to_lds_2d((unsigned int)(size_t)&Ks[0][0], ksrc, kD, kKB, kD);
      __builtin_amdgcn_s_wait_tensorcnt(0);
    } else if (w == 1) {
      tdm_load_to_lds_2d((unsigned int)(size_t)&Vs[0][0], vsrc, kD, kKB, kD);
      __builtin_amdgcn_s_wait_tensorcnt(0);
    }
#else
    h8v* kdst = (h8v*)&Ks[0][0];
    h8v* vdst = (h8v*)&Vs[0][0];
    const h8v* kvec = (const h8v*)ksrc;
    const h8v* vvec = (const h8v*)vsrc;
#pragma unroll
    for (int i = 0; i < 4; ++i) {
      kdst[tid + i * 256] = kvec[tid + i * 256];
      vdst[tid + i * 256] = vvec[tid + i * 256];
    }
#endif
    __syncthreads();

    // ---- 4 chunks of 32 keys ----
    for (int kc = 0; kc < 4; ++kc) {
      const int kk = kb * kKB + kc * 32;   // first key of chunk (global t)
      if (kk > q0 + 15) break;             // fully above diagonal
      const int kloc = kc * 32;

      // S = Q K^T : two 16x16 score tiles
      v8f s0 = {}, s1 = {};
      s0 = wmma_f16(aq0, load_frag(&Ks[kloc][0], kD, lane), s0);
      s0 = wmma_f16(aq1, load_frag(&Ks[kloc][32], kD, lane), s0);
      s1 = wmma_f16(aq0, load_frag(&Ks[kloc + 16][0], kD, lane), s1);
      s1 = wmma_f16(aq1, load_frag(&Ks[kloc + 16][32], kD, lane), s1);

      // scale + causal mask + online softmax (row = cross-lane group of 16)
#pragma unroll
      for (int r = 0; r < 8; ++r) {
        const int q = q0 + r + 8 * hi;
        float v0 = (kk + col <= q)      ? s0[r] * sc : -1e30f;
        float v1 = (kk + 16 + col <= q) ? s1[r] * sc : -1e30f;
        float mx = fmaxf(v0, v1);
        mx = fmaxf(mx, __shfl_xor(mx, 1, 32));
        mx = fmaxf(mx, __shfl_xor(mx, 2, 32));
        mx = fmaxf(mx, __shfl_xor(mx, 4, 32));
        mx = fmaxf(mx, __shfl_xor(mx, 8, 32));
        const float mn = fmaxf(m[r], mx);
        const float p0 = __expf(v0 - mn);
        const float p1 = __expf(v1 - mn);
        float rs = p0 + p1;
        rs += __shfl_xor(rs, 1, 32);
        rs += __shfl_xor(rs, 2, 32);
        rs += __shfl_xor(rs, 4, 32);
        rs += __shfl_xor(rs, 8, 32);
        const float alpha = __expf(m[r] - mn);
        m[r] = mn;
        l[r] = l[r] * alpha + rs;
#pragma unroll
        for (int nt = 0; nt < 4; ++nt) o[nt][r] *= alpha;
        // park P (f16) for re-layout as an A fragment
        Ps[w][r + 8 * hi][col] = (_Float16)p0;
        Ps[w][r + 8 * hi][16 + col] = (_Float16)p1;
      }

      // O += P @ V  (A = 16x32 P, B = 32-key slab of row-major Vs)
      v16h ap = load_frag(&Ps[w][0][0], 40, lane);
#pragma unroll
      for (int nt = 0; nt < 4; ++nt) {
        v16h bv = load_frag_kmajor(&Vs[kloc][nt * 16], kD, lane);
        o[nt] = wmma_f16(ap, bv, o[nt]);
      }
    }
  }

  // epilogue: normalize, write attention output f16 [B*T, C] (head-interleaved)
  _Float16* obase = Oh + (size_t)(b * kT + q0) * kC + hd * kD;
#pragma unroll
  for (int r = 0; r < 8; ++r) {
    const float inv = 1.0f / l[r];
#pragma unroll
    for (int nt = 0; nt < 4; ++nt) {
      obase[(size_t)(r + 8 * hi) * kC + nt * 16 + col] = (_Float16)(o[nt][r] * inv);
    }
  }
}

// ---------------------------------------------------------------------------
// Kernel 4: output projection  out = O @ Wo^T + bo  (fp32 out)
// ---------------------------------------------------------------------------
__global__ __launch_bounds__(256) void proj_kernel(
    const _Float16* __restrict__ Oh, const _Float16* __restrict__ Woh,
    const float* __restrict__ bo, float* __restrict__ out) {
  __shared__ __align__(16) _Float16 Os[16][kC];  // 12 KB

  const int tid = threadIdx.x;
  const int row0 = blockIdx.x * 16;
  const _Float16* osrc = Oh + (size_t)row0 * kC;

#if USE_TDM
  if (tid < 32) {
    // one flat 6144-element row: stage the whole 16x384 tile with one D#
    tdm_load_to_lds_2d((unsigned int)(size_t)&Os[0][0], osrc, 16 * kC, 1, 16 * kC);
    __builtin_amdgcn_s_wait_tensorcnt(0);
  }
#else
  h8v* dsts = (h8v*)&Os[0][0];
  const h8v* src = (const h8v*)osrc;
#pragma unroll
  for (int i = 0; i < 3; ++i) dsts[tid + i * 256] = src[tid + i * 256];
#endif
  __syncthreads();

  const int w = tid >> 5, lane = tid & 31;
  const int col = lane & 15, hi = lane >> 4;

  v8f acc[3] = {};
#pragma unroll
  for (int kc = 0; kc < 12; ++kc) {
    v16h a = load_frag(&Os[0][kc * 32], kC, lane);
    v16h bf[3];
#pragma unroll
    for (int s = 0; s < 3; ++s) {
      const int j0 = (w * 3 + s) * 16;
      bf[s] = load_frag(Woh + (size_t)j0 * kC + kc * 32, kC, lane);
    }
#pragma unroll
    for (int s = 0; s < 3; ++s) acc[s] = wmma_f16(a, bf[s], acc[s]);
  }

#pragma unroll
  for (int s = 0; s < 3; ++s) {
    const int j0 = (w * 3 + s) * 16;
    const float bias = bo[j0 + col];
#pragma unroll
    for (int r = 0; r < 8; ++r) {
      out[(size_t)(row0 + r + 8 * hi) * kC + j0 + col] = acc[s][r] + bias;
    }
  }
}

// ---------------------------------------------------------------------------
extern "C" void kernel_launch(void* const* d_in, const int* in_sizes, int n_in,
                              void* d_out, int out_size, void* d_ws, size_t ws_size,
                              hipStream_t stream) {
  (void)in_sizes; (void)n_in; (void)out_size; (void)ws_size;

  const float* x  = (const float*)d_in[0];
  const float* Wk = (const float*)d_in[1];
  const float* Wq = (const float*)d_in[2];
  const float* Wv = (const float*)d_in[3];
  const float* Wo = (const float*)d_in[4];
  const float* bo = (const float*)d_in[5];
  float* out = (float*)d_out;

  // workspace layout (f16): [Wk|Wq|Wv|Wo] weights, then Q, K, V, O
  char* ws = (char*)d_ws;
  const size_t wbytes = (size_t)4 * kC * kC * sizeof(_Float16);   // 1.18 MB
  const size_t tbytes = (size_t)kBT * kC * sizeof(_Float16);      // 12.6 MB each
  _Float16* Wh = (_Float16*)ws;
  _Float16* Qh = (_Float16*)(ws + wbytes);
  _Float16* Kh = (_Float16*)(ws + wbytes + tbytes);
  _Float16* Vh = (_Float16*)(ws + wbytes + 2 * tbytes);
  _Float16* Oh = (_Float16*)(ws + wbytes + 3 * tbytes);

  cvt_weights_kernel<<<(4 * kC * kC + 255) / 256, 256, 0, stream>>>(
      Wk, Wq, Wv, Wo, Wh);
  qkv_kernel<<<kBT / 16, 256, 0, stream>>>(x, Wh, Qh, Kh, Vh);
  attn_kernel<<<kB * kH * 2, 256, 0, stream>>>(Qh, Kh, Vh, Oh);
  proj_kernel<<<kBT / 16, 256, 0, stream>>>(Oh, Wh + 3 * kC * kC, bo, out);
}
